// Model_27917287424706
// MI455X (gfx1250) — matrix operand order repaired and checked by reading, not compiled, not run on hardware
//
#include <hip/hip_runtime.h>

typedef __attribute__((ext_vector_type(16))) __bf16       v16bf;
typedef __attribute__((ext_vector_type(8)))  float        v8f;
typedef __attribute__((ext_vector_type(8)))  unsigned int u32x8;
typedef __attribute__((ext_vector_type(4)))  unsigned int u32x4;
typedef __attribute__((ext_vector_type(8)))  int          i32x8;
typedef __attribute__((ext_vector_type(4)))  int          i32x4;
typedef __attribute__((ext_vector_type(4)))  float        f32x4;

#define B_SZ   256
#define T_SZ   1024
#define IN_SZ  64
#define K_SZ   128
#define G4     512
#define M_OUT  63

// LDS layout (byte offsets), total < 320KB per WGP (CDNA5)
#define WXB_OFF   0u        // Wx^T B-fragments: 2 kb x 32 nb x 1KB = 64KB
#define UHB_OFF   65536u    // Uh^T B-fragments: 4 kb x 32 nb x 1KB = 128KB
#define VWB_OFF   196608u   // Vw^T B-fragments: 4 kb x 4 nb x 1KB = 16KB
#define ZBUF_OFF  212992u   // z scratch: 16 rows x 520 f32 = 33280B (padded stride)
#define HLDS_OFF  246272u   // h A-fragments: 4 kb x 1KB = 4KB
#define XFRG_OFF  250368u   // x A-fragments, double buffered: 2 x 2KB
#define XRAW_OFF  254464u   // raw f32 x tiles (TDM dest), double buffered: 2 x 4KB
#define LDS_TOTAL 262656u
#define ZSTR 520u

#define HAVE_TDM __has_builtin(__builtin_amdgcn_tensor_load_to_lds)

__device__ __forceinline__ unsigned short f2bf(float f) {
    unsigned int u = __builtin_bit_cast(unsigned int, f);
    unsigned int r = (u + 0x7FFFu + ((u >> 16) & 1u)) >> 16;
    return (unsigned short)r;
}

// Fast gate activations: only v_exp_f32 (exp2) + v_rcp_f32 trans ops, which
// co-execute with XDL WMMA (16-bit WMMA is TRANS-class on CDNA5).
__device__ __forceinline__ float fsig(float v) {   // 1/(1+e^-v)
    return __builtin_amdgcn_rcpf(1.f + __builtin_amdgcn_exp2f(-1.44269504089f * v));
}
__device__ __forceinline__ float ftanh(float v) {  // 1 - 2/(e^{2v}+1)
    return 1.f - 2.f * __builtin_amdgcn_rcpf(1.f + __builtin_amdgcn_exp2f(2.88539008178f * v));
}

// B-matrix (K x N) fragment address: 32B per lane, fragment-major.
__device__ __forceinline__ unsigned bfragAddr(unsigned region, unsigned NB,
                                              unsigned kb, unsigned nb,
                                              unsigned k32, unsigned n16) {
    unsigned lane_hi = (k32 >> 3) & 1u;
    unsigned kk      = ((k32 & 16u) >> 1) + (k32 & 7u);
    unsigned lane    = n16 + (lane_hi << 4);
    return region + ((kb * NB + nb) << 10) + (lane << 5) + ((kk >> 1) << 2) + ((kk & 1u) << 1);
}

// A-matrix (16 x 32, 16-bit) fragment address per ISA table.
__device__ __forceinline__ unsigned afragAddr(unsigned region, unsigned kb,
                                              unsigned m, unsigned k32) {
    unsigned lane = m + (((k32 >> 3) & 1u) << 4);
    unsigned v    = ((k32 & 16u) >> 2) + ((k32 & 7u) >> 1);
    return region + (kb << 10) + (lane << 5) + (v << 2) + ((k32 & 1u) << 1);
}

__device__ __forceinline__ v16bf ldsFrag(const char* p) {
    union { u32x8 u; v16bf b; } c;
    c.u = *(const u32x8*)p;     // 2x ds_load_b128
    return c.b;
}

__global__ void __launch_bounds__(256)
lstm_persist(const float* __restrict__ x,
             const float* __restrict__ Wi, const float* __restrict__ Ui,
             const float* __restrict__ Wf, const float* __restrict__ Uf,
             const float* __restrict__ Wg, const float* __restrict__ Ug,
             const float* __restrict__ Wo, const float* __restrict__ Uo,
             const float* __restrict__ Vw, const float* __restrict__ Vb,
             float* __restrict__ out)
{
    extern __shared__ char lds[];
    const unsigned tid  = threadIdx.x;
    const unsigned lane = tid & 31u;
    const unsigned wv   = (unsigned)__builtin_amdgcn_readfirstlane((int)(tid >> 5)); // uniform
    const unsigned b0   = blockIdx.x * 16u;

    // ---------------- one-time weight staging into swizzled LDS (bf16) -----------
    for (unsigned i = tid; i < (16384u >> 2); i += 256u) ((unsigned*)(lds + VWB_OFF))[i] = 0u;
    for (unsigned i = tid; i < (4096u  >> 2); i += 256u) ((unsigned*)(lds + HLDS_OFF))[i] = 0u; // h0 = 0

    for (unsigned idx = tid; idx < G4 * IN_SZ; idx += 256u) {
        unsigned j = idx >> 6, k = idx & 63u;
        unsigned g = j >> 7, r = j & 127u;
        const float* Wp = (g == 0u) ? Wi : (g == 1u) ? Wf : (g == 2u) ? Wg : Wo;
        float v = Wp[r * IN_SZ + k];
        *(unsigned short*)(lds + bfragAddr(WXB_OFF, 32u, k >> 5, j >> 4, k & 31u, j & 15u)) = f2bf(v);
    }
    for (unsigned idx = tid; idx < G4 * K_SZ; idx += 256u) {
        unsigned j = idx >> 7, k = idx & 127u;
        unsigned g = j >> 7, r = j & 127u;
        const float* Up = (g == 0u) ? Ui : (g == 1u) ? Uf : (g == 2u) ? Ug : Uo;
        float v = Up[r * K_SZ + k];
        *(unsigned short*)(lds + bfragAddr(UHB_OFF, 32u, k >> 5, j >> 4, k & 31u, j & 15u)) = f2bf(v);
    }
    for (unsigned idx = tid; idx < (unsigned)(M_OUT * K_SZ); idx += 256u) {
        unsigned m = idx >> 7, k = idx & 127u;
        float v = Vw[m * K_SZ + k];
        *(unsigned short*)(lds + bfragAddr(VWB_OFF, 4u, k >> 5, m >> 4, k & 31u, m & 15u)) = f2bf(v);
    }
    // Stage x_0 straight into A-fragment layout (bf16), buffer 0
    for (unsigned i = 0; i < 4u; ++i) {
        unsigned e = tid + i * 256u;
        unsigned row = e >> 6, col = e & 63u;
        float v = x[(b0 + row) * (T_SZ * IN_SZ) + col];
        *(unsigned short*)(lds + XFRG_OFF + afragAddr(0u, col >> 5, row, col & 31u)) = f2bf(v);
    }

    // Per-thread recurrent cell state: c for (row = tid>>4, k = (tid&15)*8 + q)
    const unsigned erow  = tid >> 4;
    const unsigned kbase = (tid & 15u) * 8u;
    float creg[8];
#pragma unroll
    for (int q = 0; q < 8; ++q) creg[q] = 0.f;

    float biasv = 0.f;
    if (wv < 4u) {
        unsigned n = wv * 16u + (lane & 15u);
        if (n < M_OUT) biasv = Vb[n];
    }

    __syncthreads();

    // ---------------- recurrence over T -----------------------------------------
    for (unsigned t = 0; t < T_SZ; ++t) {
#if HAVE_TDM
        // Kick the Tensor Data Mover for x_{t+1}: 16x64 f32 tile, row stride T*IN.
        // DMA runs while this wave does the gate GEMM below. TENSORcnt is per-wave,
        // so wave 4 both issues and waits; other waves see the data only through
        // wave 4's bf16 conversion stores + the phase barrier.
        if (wv == 4u && t + 1u < T_SZ) {
            unsigned ldsDst = XRAW_OFF + (((t + 1u) & 1u) << 12);
            unsigned long long ga = (unsigned long long)(uintptr_t)x +
                (unsigned long long)(b0 * (T_SZ * IN_SZ) + (t + 1u) * IN_SZ) * 4ull;
            u32x4 g0;
            g0[0] = 1u;                                        // count=1, user mode
            g0[1] = ldsDst;                                    // lds_addr
            g0[2] = (unsigned)ga;                              // global_addr[31:0]
            g0[3] = ((unsigned)(ga >> 32) & 0x01FFFFFFu) | (2u << 30); // addr[56:32], type=2
            i32x8 g1;
            g1[0] = 0x00020000;                                // data_size=4B, no mask
            g1[1] = (int)((unsigned)IN_SZ << 16);              // tensor_dim0=64 (lo16)
            g1[2] = (int)(16u << 16);                          // dim0 hi=0 | tensor_dim1=16
            g1[3] = (int)((unsigned)IN_SZ << 16);              // tile_dim0=64
            g1[4] = 16;                                        // tile_dim1=16, tile_dim2=0
            g1[5] = (int)(T_SZ * IN_SZ);                       // tensor_dim0_stride=65536
            g1[6] = 0;
            g1[7] = 0;
            i32x4 z4 = {0, 0, 0, 0};
            i32x8 z8 = {0, 0, 0, 0, 0, 0, 0, 0};
            // clang-23 / therock-10.0 form: 6 args (g0, g1, g2, g3, extra, cpol)
            __builtin_amdgcn_tensor_load_to_lds(g0, g1, z4, z4, z8, 0);
        }
#endif
        const char* xbuf = lds + XFRG_OFF + ((t & 1u) << 11);

        // A fragments (x: 2 kb, h: 4 kb) — shared across this wave's 4 n-tiles
        v16bf xa0 = ldsFrag(xbuf + (0u << 10) + (lane << 5));
        v16bf xa1 = ldsFrag(xbuf + (1u << 10) + (lane << 5));
        v16bf ha[4];
#pragma unroll
        for (unsigned kb = 0; kb < 4u; ++kb)
            ha[kb] = ldsFrag(lds + HLDS_OFF + (kb << 10) + (lane << 5));

        // z = x_t @ Wx^T + h_prev @ Uh^T   (each wave: 64 of the 512 gate columns)
        float* zb = (float*)(lds + ZBUF_OFF);
#pragma unroll
        for (unsigned nt = 0; nt < 4u; ++nt) {
            unsigned nb = wv * 4u + nt;
            v8f acc = {};
            {
                v16bf bf0 = ldsFrag(lds + WXB_OFF + ((0u * 32u + nb) << 10) + (lane << 5));
                acc = __builtin_amdgcn_wmma_f32_16x16x32_bf16(false, xa0, false, bf0, (short)0, acc, false, false);
                v16bf bf1 = ldsFrag(lds + WXB_OFF + ((1u * 32u + nb) << 10) + (lane << 5));
                acc = __builtin_amdgcn_wmma_f32_16x16x32_bf16(false, xa1, false, bf1, (short)0, acc, false, false);
            }
#pragma unroll
            for (unsigned kb = 0; kb < 4u; ++kb) {
                v16bf bfr = ldsFrag(lds + UHB_OFF + ((kb * 32u + nb) << 10) + (lane << 5));
                acc = __builtin_amdgcn_wmma_f32_16x16x32_bf16(false, ha[kb], false, bfr, (short)0, acc, false, false);
            }
            unsigned n    = nb * 16u + (lane & 15u);
            unsigned mrow = (lane >> 4) * 8u;
#pragma unroll
            for (int v = 0; v < 8; ++v) zb[(mrow + (unsigned)v) * ZSTR + n] = acc[v];
        }
        __syncthreads();

        // Elementwise cell update: c = f*c_prev + i*g ; h = o*tanh(c_prev)
#pragma unroll
        for (int q = 0; q < 8; ++q) {
            unsigned k  = kbase + (unsigned)q;
            float zi = zb[erow * ZSTR + k];
            float zf = zb[erow * ZSTR + 128u + k];
            float zg = zb[erow * ZSTR + 256u + k];
            float zo = zb[erow * ZSTR + 384u + k];
            float ig = fsig(zi), fg = fsig(zf), gg = ftanh(zg), og = fsig(zo);
            float cp = creg[q];
            creg[q]  = fg * cp + ig * gg;
            float h  = og * ftanh(cp);              // reference uses c_prev here
            *(unsigned short*)(lds + afragAddr(HLDS_OFF, k >> 5, erow, k & 31u)) = f2bf(h);
        }
        __syncthreads();

        if (wv < 4u) {
            // y = h @ Vw^T + Vb   (waves 0..3, one 16-col tile each)
            v8f acc;
#pragma unroll
            for (int v = 0; v < 8; ++v) acc[v] = biasv;
#pragma unroll
            for (unsigned kb = 0; kb < 4u; ++kb) {
                v16bf hf = ldsFrag(lds + HLDS_OFF + (kb << 10) + (lane << 5));
                v16bf bf = ldsFrag(lds + VWB_OFF + ((kb * 4u + wv) << 10) + (lane << 5));
                acc = __builtin_amdgcn_wmma_f32_16x16x32_bf16(false, hf, false, bf, (short)0, acc, false, false);
            }
            unsigned n = wv * 16u + (lane & 15u);
            if (n < M_OUT) {
                unsigned mrow = (lane >> 4) * 8u;
#pragma unroll
                for (int v = 0; v < 8; ++v) {
                    unsigned o = ((b0 + mrow + (unsigned)v) * T_SZ + t) * M_OUT + n;
                    out[o] = acc[v];
                }
            }
        }
#if HAVE_TDM
        else if (wv == 4u) {
            if (t + 1u < T_SZ) {
                // Wait for the TDM tile, then convert raw f32 -> bf16 A-fragments.
                __builtin_amdgcn_s_wait_tensorcnt(0);
                const char* xraw = lds + XRAW_OFF + (((t + 1u) & 1u) << 12);
                char*       xfrg = lds + XFRG_OFF + (((t + 1u) & 1u) << 11);
                unsigned row = lane >> 1, kb = lane & 1u, cb = kb << 5;
#pragma unroll
                for (unsigned j4 = 0; j4 < 32u; j4 += 4u) {
                    f32x4 v = *(const f32x4*)(xraw + ((row * IN_SZ + cb + j4) << 2));
#pragma unroll
                    for (unsigned p = 0; p < 4u; p += 2u) {
                        unsigned k32 = cb + j4 + p;    // even: pairs share a DWORD
                        unsigned w = (unsigned)f2bf(v[p]) | ((unsigned)f2bf(v[p + 1u]) << 16);
                        *(unsigned*)(xfrg + afragAddr(0u, kb, row, k32 & 31u)) = w;
                    }
                }
            }
        } else if (wv == 5u) {
            if (t + 2u < T_SZ)   // warm L2 for the tile after next (global_prefetch_b8)
                __builtin_prefetch(x + b0 * (T_SZ * IN_SZ) + (t + 2u) * IN_SZ, 0, 1);
        }
#else
        else if (t + 1u < T_SZ) {
            // Fallback: waves 4..7 stage x_{t+1} with plain loads
            unsigned tid2  = tid - 128u;
            char* nxbuf = lds + XFRG_OFF + (((t + 1u) & 1u) << 11);
#pragma unroll
            for (unsigned i = 0; i < 8u; ++i) {
                unsigned e = tid2 + i * 128u;
                unsigned row = e >> 6, col = e & 63u;
                float v = x[(b0 + row) * (T_SZ * IN_SZ) + (t + 1u) * IN_SZ + col];
                *(unsigned short*)(nxbuf + afragAddr(0u, col >> 5, row, col & 31u)) = f2bf(v);
            }
        }
#endif
        __syncthreads();
    }
}

extern "C" void kernel_launch(void* const* d_in, const int* in_sizes, int n_in,
                              void* d_out, int out_size, void* d_ws, size_t ws_size,
                              hipStream_t stream) {
    const float* x  = (const float*)d_in[0];
    const float* Wi = (const float*)d_in[1];
    const float* Ui = (const float*)d_in[2];
    const float* Wf = (const float*)d_in[3];
    const float* Uf = (const float*)d_in[4];
    const float* Wg = (const float*)d_in[5];
    const float* Ug = (const float*)d_in[6];
    const float* Wo = (const float*)d_in[7];
    const float* Uo = (const float*)d_in[8];
    const float* Vw = (const float*)d_in[9];
    const float* Vb = (const float*)d_in[10];
    float* out = (float*)d_out;

    (void)hipFuncSetAttribute(reinterpret_cast<const void*>(lstm_persist),
                              hipFuncAttributeMaxDynamicSharedMemorySize, (int)LDS_TOTAL);
    lstm_persist<<<B_SZ / 16, 256, LDS_TOTAL, stream>>>(x, Wi, Ui, Wf, Uf, Wg, Ug, Wo, Uo, Vw, Vb, out);
}
